// AnalogMultiheadAttention_43696997269960
// MI455X (gfx1250) — matrix operand order repaired and checked by reading, not compiled, not run on hardware
//
#include <hip/hip_runtime.h>

#if __has_include(<hip/amd_detail/amd_gfx1250_TDM.h>)
#define TDM_SIX_ARG 1
#else
#define TDM_SIX_ARG 0
#endif

typedef __attribute__((ext_vector_type(16))) _Float16     v16h;
typedef __attribute__((ext_vector_type(8)))  float        v8f;
typedef __attribute__((ext_vector_type(4)))  unsigned int v4u;
typedef __attribute__((ext_vector_type(8)))  int          v8i;
typedef __attribute__((ext_vector_type(4)))  int          v4i;

#define E_DIM 768
#define H_DIM 12
#define D_DIM 64
#define B_DIM 4
#define T_DIM 2048
#define M_DIM (B_DIM * T_DIM)      // 8192 rows through every projection

#define SIGMA_TH   6.4357964e-5f   // sqrt(kT/C)
#define QSTEP      (2.0f / 255.0f)
#define INV_QSTEP  127.5f

// ---------------------------------------------------------------------------
// WMMA fragment loaders (wave32; contiguous k = 16*(l/16) + j, same order in A and B).
// Work on both global and LDS pointers (forceinline -> addrspace inference).
// ---------------------------------------------------------------------------

// A-matrix 16x32 f16: lane l holds row M=l%16.
// half j -> K = j + 16*(l/16) (matches B loaders)
static __device__ __forceinline__ v16h load_frag_a(const _Float16* __restrict__ base, int ld) {
    const int lane = threadIdx.x & 31;
    const int row  = lane & 15;
    const int kh   = (lane >> 4) << 4;          // 0 or 16
    const _Float16* p = base + (size_t)row * ld + kh;
    v16h a;
#pragma unroll
    for (int j = 0; j < 16; ++j) a[j] = p[j];
    return a;
}

// B-matrix 32x16 f16 where memory is [N][K] row-major (e.g. W[n,k], K[t,d]):
// lane l holds column N=l%16; half j -> K = j + 16*(l/16). 32B contiguous per lane.
static __device__ __forceinline__ v16h load_frag_b_nk(const _Float16* __restrict__ base, int ld) {
    const int lane = threadIdx.x & 31;
    const int n  = lane & 15;
    const int kb = (lane >> 4) << 4;            // 0 or 16
    const _Float16* p = base + (size_t)n * ld + kb;
    v16h b;
#pragma unroll
    for (int j = 0; j < 16; ++j) b[j] = p[j];
    return b;
}

// B-matrix 32x16 f16 where memory is [K][N] row-major (e.g. V[t,d] with K=t):
static __device__ __forceinline__ v16h load_frag_b_kn(const _Float16* __restrict__ base, int ld) {
    const int lane = threadIdx.x & 31;
    const int n  = lane & 15;
    const int kb = (lane >> 4) << 4;
    v16h b;
#pragma unroll
    for (int j = 0; j < 16; ++j) b[j] = base[(size_t)(kb + j) * ld + n];
    return b;
}

static __device__ __forceinline__ v8f wmma_f16(v16h a, v16h b, v8f c) {
    return __builtin_amdgcn_wmma_f32_16x16x32_f16(false, a, false, b, (short)0, c, false, false);
}

// 16-lane (row-group) reductions for the C/D layout (N = lane%16 within each half)
static __device__ __forceinline__ float rmax16(float v) {
#pragma unroll
    for (int off = 8; off; off >>= 1) v = fmaxf(v, __shfl_xor(v, off, 16));
    return v;
}
static __device__ __forceinline__ float rsum16(float v) {
#pragma unroll
    for (int off = 8; off; off >>= 1) v += __shfl_xor(v, off, 16);
    return v;
}

// ---------------------------------------------------------------------------
// Tensor Data Mover: 2D tile (global, 16-bit elements) -> LDS.
// D# packing per CDNA5 ISA 8.3/8.4:
//   group0: count=1 | lds_addr | global_addr[56:0] | type=2
//   group1: data_size=1(2B), tensor_dim0/1, tile_dim0/1, tensor_dim0_stride
// tile_d0 = contiguous (K) extent, tile_d1 = number of rows.
// ---------------------------------------------------------------------------
static __device__ __forceinline__ void tdm_load_2d_f16(
        const _Float16* lds_dst, const _Float16* gsrc,
        unsigned tensor_d0, unsigned tensor_d1, unsigned stride_d0,
        unsigned tile_d0, unsigned tile_d1) {
    const unsigned long long ga = (unsigned long long)(const void*)gsrc;  // byte addr
    const unsigned lds_addr = (unsigned)(size_t)(const void*)lds_dst;     // LDS byte offset

    v4u g0;
    g0[0] = 1u;                                           // count=1, user descriptor
    g0[1] = lds_addr;                                     // lds_addr [63:32]
    g0[2] = (unsigned)(ga & 0xffffffffu);                 // global_addr [95:64]
    g0[3] = (unsigned)((ga >> 32) & 0x01ffffffu)          // global_addr [120:96]
          | (2u << 30);                                   // type=2 ("image")

    v8i g1;
    g1[0] = (int)(1u << 16);                              // wg_mask=0, data_size=1 (2B)
    g1[1] = (int)((tensor_d0 & 0xffffu) << 16);           // tensor_dim0[15:0]
    g1[2] = (int)((tensor_d0 >> 16) | ((tensor_d1 & 0xffffu) << 16));
    g1[3] = (int)((tensor_d1 >> 16) | ((tile_d0 & 0xffffu) << 16));   // tile_dim0
    g1[4] = (int)(tile_d1 & 0xffffu);                     // tile_dim1 (tile_dim2=0)
    g1[5] = (int)stride_d0;                               // tensor_dim0_stride[31:0]
    g1[6] = 0;                                            // stride hi / dim1_stride lo
    g1[7] = 0;

    v4i z4 = {0, 0, 0, 0};
#if TDM_SIX_ARG
    v8i z8 = {0, 0, 0, 0, 0, 0, 0, 0};
    __builtin_amdgcn_tensor_load_to_lds(g0, g1, z4, z4, z8, 0);
#else
    __builtin_amdgcn_tensor_load_to_lds(g0, g1, z4, z4, 0);
#endif
}

// Deterministic hash -> N(0,1). (JAX PRNG streams are not bit-reproducible here;
// sigma_th is ~100x below the ADC quantization step, so any iid normal is faithful.)
static __device__ __forceinline__ float gauss_hash(unsigned idx, unsigned seed) {
    unsigned s = idx ^ (seed * 0x9E3779B9u);
    s = s * 747796405u + 2891336453u;
    unsigned w1 = ((s >> ((s >> 28) + 4u)) ^ s) * 277803737u; w1 = (w1 >> 22) ^ w1;
    s = s * 747796405u + 2891336453u;
    unsigned w2 = ((s >> ((s >> 28) + 4u)) ^ s) * 277803737u; w2 = (w2 >> 22) ^ w2;
    float u1 = (float)(w1 + 1u) * 2.3283064365386963e-10f;   // (0,1]
    float u2 = (float)w2        * 2.3283064365386963e-10f;   // [0,1)
    float r = __fsqrt_rn(-2.0f * __logf(u1));
    float sn, cs;
    __sincosf(6.28318530717958647f * u2, &sn, &cs);
    return r * cs;
}

// ---------------------------------------------------------------------------
// Prep kernels
// ---------------------------------------------------------------------------
__global__ void weff_kernel(const float* __restrict__ W, const float* __restrict__ delta,
                            _Float16* __restrict__ Wh, int n) {
    int i = blockIdx.x * blockDim.x + threadIdx.x;
    if (i < n) Wh[i] = (_Float16)(W[i] * (1.0f + delta[i]));
}

__global__ void f32_to_f16_kernel(const float* __restrict__ src, _Float16* __restrict__ dst, int n) {
    int i = blockIdx.x * blockDim.x + threadIdx.x;
    if (i < n) dst[i] = (_Float16)src[i];
}

// ---------------------------------------------------------------------------
// Analog GEMM:  Y[m,n] = ADC( X[m,:] . Weff[n,:] + b[n] + noise )
//  block = 256 threads (8 waves); block tile 64(M) x 128(N); wave tile 32x32.
//  A/B tiles staged to LDS by the Tensor Data Mover, double-buffered:
//  each wave DMAs its own slice (8 A-rows, 16 B-rows) and pipelines with
//  s_wait_tensorcnt <= 2 while the next stage's two descriptors are in flight.
// ---------------------------------------------------------------------------
__global__ void __launch_bounds__(256)
analog_gemm_kernel(const _Float16* __restrict__ X, const _Float16* __restrict__ W,
                   const float* __restrict__ bias,
                   float* __restrict__ Yf32, _Float16* __restrict__ Yf16,
                   float f16_scale, unsigned seed) {
    __shared__ _Float16 As[2][64 * 32];    // [buf][m][k]
    __shared__ _Float16 Bs[2][128 * 32];   // [buf][n][k]

    const int wave = threadIdx.x >> 5;     // 0..7
    const int wm   = wave >> 2;            // 0..1  (32-row slab)
    const int wn   = wave & 3;             // 0..3  (32-col slab)
    const int m0 = blockIdx.x * 64;
    const int n0 = blockIdx.y * 128;

    // per-wave DMA slices
    const _Float16* gA = X + (size_t)(m0 + wave * 8)  * E_DIM;
    const _Float16* gB = W + (size_t)(n0 + wave * 16) * E_DIM;
    const unsigned ldsA = wave * 8  * 32;  // halves offset inside buffer
    const unsigned ldsB = wave * 16 * 32;

#define TDM_STAGE(buf, kk)                                                              \
    do {                                                                                \
        tdm_load_2d_f16(&As[buf][ldsA], gA + (kk), E_DIM, M_DIM, E_DIM, 32u, 8u);       \
        tdm_load_2d_f16(&Bs[buf][ldsB], gB + (kk), E_DIM, E_DIM, E_DIM, 32u, 16u);      \
    } while (0)

    v8f a00 = {}, a01 = {}, a10 = {}, a11 = {};

    TDM_STAGE(0, 0);
    for (int it = 0; it < E_DIM / 32; ++it) {
        if (it < E_DIM / 32 - 1) {
            TDM_STAGE((it + 1) & 1, (it + 1) * 32);
            __builtin_amdgcn_s_wait_tensorcnt(2);   // stage `it` (2 descriptors) done
        } else {
            __builtin_amdgcn_s_wait_tensorcnt(0);
        }
        __syncthreads();                            // all waves' slices visible

        const int buf = it & 1;
        const v16h fa0 = load_frag_a(&As[buf][(wm * 32 + 0)  * 32], 32);
        const v16h fa1 = load_frag_a(&As[buf][(wm * 32 + 16) * 32], 32);
        const v16h fb0 = load_frag_b_nk(&Bs[buf][(wn * 32 + 0)  * 32], 32);
        const v16h fb1 = load_frag_b_nk(&Bs[buf][(wn * 32 + 16) * 32], 32);
        a00 = wmma_f16(fa0, fb0, a00);
        a01 = wmma_f16(fa0, fb1, a01);
        a10 = wmma_f16(fa1, fb0, a10);
        a11 = wmma_f16(fa1, fb1, a11);

        __syncthreads();                            // safe to overwrite buf next round
    }
#undef TDM_STAGE

    // ---- ADC epilogue: bias + thermal noise + clip + 8-bit quantize ----
    const int lane = threadIdx.x & 31;
    const int colN = lane & 15;
    const int rowH = (lane >> 4) << 3;
    const v8f accs[2][2] = {{a00, a01}, {a10, a11}};
#pragma unroll
    for (int mi = 0; mi < 2; ++mi) {
#pragma unroll
        for (int ni = 0; ni < 2; ++ni) {
            const v8f acc = accs[mi][ni];
            const int nn = n0 + wn * 32 + ni * 16 + colN;
            const float bn = bias[nn];
#pragma unroll
            for (int r = 0; r < 8; ++r) {
                const int mm = m0 + wm * 32 + mi * 16 + r + rowH;
                float y = acc[r] + bn;
                y += SIGMA_TH * gauss_hash((unsigned)(mm * E_DIM + nn), seed);
                float yc = fminf(1.0f, fmaxf(-1.0f, y));
                float yq = rintf(yc * INV_QSTEP) * QSTEP;   // RNE matches jnp.round
                const size_t o = (size_t)mm * E_DIM + nn;
                if (Yf32) Yf32[o] = yq;
                if (Yf16) Yf16[o] = (_Float16)(yq * f16_scale);
            }
        }
    }
}

// ---------------------------------------------------------------------------
// Attention: block = (b, 16 query rows), 12 waves = 12 heads.
// Two-pass streaming softmax; head-mean reduced in LDS (deterministic, no atomics).
// Q already carries the D^-1/2 scale (folded at projection time).
// ---------------------------------------------------------------------------
__global__ void __launch_bounds__(384)
attn_kernel(const _Float16* __restrict__ Qh, const _Float16* __restrict__ Kh,
            const _Float16* __restrict__ Vh, _Float16* __restrict__ Oh,
            float* __restrict__ attn_mean) {
    __shared__ _Float16 Pf16[H_DIM][16 * 32];
    __shared__ float    Pf32[H_DIM][16 * 32];

    const int h    = threadIdx.x >> 5;       // head = wave
    const int lane = threadIdx.x & 31;
    const int qt   = blockIdx.x;             // query tile (16 rows)
    const int b    = blockIdx.y;
    const size_t rowbase = (size_t)b * T_DIM + qt * 16;

    const _Float16* Qp = Qh + rowbase * E_DIM + h * D_DIM;
    const _Float16* Kb = Kh + (size_t)b * T_DIM * E_DIM + h * D_DIM;
    const _Float16* Vb = Vh + (size_t)b * T_DIM * E_DIM + h * D_DIM;

    const v16h aq0 = load_frag_a(Qp, E_DIM);
    const v16h aq1 = load_frag_a(Qp + 32, E_DIM);

    // ---- pass 1: running row max / sum-exp over all keys ----
    float mrow[8], lrow[8];
#pragma unroll
    for (int r = 0; r < 8; ++r) { mrow[r] = -1e30f; lrow[r] = 0.0f; }

    for (int kt = 0; kt < T_DIM; kt += 16) {
        v8f s = {};
        const _Float16* kp = Kb + (size_t)kt * E_DIM;
        s = wmma_f16(aq0, load_frag_b_nk(kp, E_DIM), s);
        s = wmma_f16(aq1, load_frag_b_nk(kp + 32, E_DIM), s);
#pragma unroll
        for (int r = 0; r < 8; ++r) {
            float v = s[r];
            float mnew = fmaxf(mrow[r], rmax16(v));
            lrow[r] = lrow[r] * __expf(mrow[r] - mnew) + rsum16(__expf(v - mnew));
            mrow[r] = mnew;
        }
    }
    float linv[8];
#pragma unroll
    for (int r = 0; r < 8; ++r) linv[r] = 1.0f / lrow[r];

    // ---- pass 2: probabilities -> (PV via WMMA) + head-mean write ----
    v8f o0 = {}, o1 = {}, o2 = {}, o3 = {};
    const int n = lane & 15;

    for (int kt = 0; kt < T_DIM; kt += 32) {
        const _Float16* kp = Kb + (size_t)kt * E_DIM;
        v8f s0 = {}, s1 = {};
        s0 = wmma_f16(aq0, load_frag_b_nk(kp, E_DIM), s0);
        s0 = wmma_f16(aq1, load_frag_b_nk(kp + 32, E_DIM), s0);
        s1 = wmma_f16(aq0, load_frag_b_nk(kp + (size_t)16 * E_DIM, E_DIM), s1);
        s1 = wmma_f16(aq1, load_frag_b_nk(kp + (size_t)16 * E_DIM + 32, E_DIM), s1);

#pragma unroll
        for (int r = 0; r < 8; ++r) {
            const int row = r + ((lane >> 4) << 3);
            float p0 = __expf(s0[r] - mrow[r]) * linv[r];
            float p1 = __expf(s1[r] - mrow[r]) * linv[r];
            Pf16[h][row * 32 + n]      = (_Float16)p0;
            Pf16[h][row * 32 + 16 + n] = (_Float16)p1;
            Pf32[h][row * 32 + n]      = p0;
            Pf32[h][row * 32 + 16 + n] = p1;
        }
        __syncthreads();

        // P(16x32) @ V(32x64) -> four 16x16 WMMAs
        const v16h ap = load_frag_a(&Pf16[h][0], 32);
        const _Float16* vp = Vb + (size_t)kt * E_DIM;
        o0 = wmma_f16(ap, load_frag_b_kn(vp,      E_DIM), o0);
        o1 = wmma_f16(ap, load_frag_b_kn(vp + 16, E_DIM), o1);
        o2 = wmma_f16(ap, load_frag_b_kn(vp + 32, E_DIM), o2);
        o3 = wmma_f16(ap, load_frag_b_kn(vp + 48, E_DIM), o3);

        // mean over the 12 heads: 16x32 elements, all 384 threads cooperate
        for (int idx = threadIdx.x; idx < 512; idx += 384) {
            float sum = 0.0f;
#pragma unroll
            for (int hh = 0; hh < H_DIM; ++hh) sum += Pf32[hh][idx];
            const int row = idx >> 5, col = idx & 31;
            attn_mean[(rowbase + row) * (size_t)T_DIM + kt + col] = sum * (1.0f / H_DIM);
        }
        __syncthreads();
    }

    // store O tile (f16 input to output projection), head h -> columns h*64..h*64+63
#pragma unroll
    for (int r = 0; r < 8; ++r) {
        const int row = r + ((lane >> 4) << 3);
        const size_t o = (rowbase + row) * (size_t)E_DIM + h * D_DIM;
        Oh[o + n]      = (_Float16)o0[r];
        Oh[o + 16 + n] = (_Float16)o1[r];
        Oh[o + 32 + n] = (_Float16)o2[r];
        Oh[o + 48 + n] = (_Float16)o3[r];
    }
}

// ---------------------------------------------------------------------------
// Host-side orchestration
// ---------------------------------------------------------------------------
extern "C" void kernel_launch(void* const* d_in, const int* in_sizes, int n_in,
                              void* d_out, int out_size, void* d_ws, size_t ws_size,
                              hipStream_t stream) {
    (void)in_sizes; (void)n_in; (void)out_size; (void)ws_size;

    const float* xq = (const float*)d_in[0];
    const float* xk = (const float*)d_in[1];
    const float* xv = (const float*)d_in[2];
    const float* Wq = (const float*)d_in[3];  const float* bq = (const float*)d_in[4];  const float* dq = (const float*)d_in[5];
    const float* Wk = (const float*)d_in[6];  const float* bk = (const float*)d_in[7];  const float* dk = (const float*)d_in[8];
    const float* Wv = (const float*)d_in[9];  const float* bv = (const float*)d_in[10]; const float* dv = (const float*)d_in[11];
    const float* Wo = (const float*)d_in[12]; const float* bo = (const float*)d_in[13]; const float* dO = (const float*)d_in[14];

    float* out       = (float*)d_out;
    float* attn_mean = out + (size_t)M_DIM * E_DIM;   // out is (B,T,E) first

    // workspace carve-up (f16 staging, ~89 MB total)
    const size_t szW = (size_t)E_DIM * E_DIM * sizeof(_Float16);
    const size_t szX = (size_t)M_DIM * E_DIM * sizeof(_Float16);
    char* ws = (char*)d_ws;
    _Float16* WqH = (_Float16*)(ws);             ws += szW;
    _Float16* WkH = (_Float16*)(ws);             ws += szW;
    _Float16* WvH = (_Float16*)(ws);             ws += szW;
    _Float16* WoH = (_Float16*)(ws);             ws += szW;
    _Float16* XqH = (_Float16*)(ws);             ws += szX;
    _Float16* XkH = (_Float16*)(ws);             ws += szX;
    _Float16* XvH = (_Float16*)(ws);             ws += szX;
    _Float16* Qh  = (_Float16*)(ws);             ws += szX;
    _Float16* Kh  = (_Float16*)(ws);             ws += szX;
    _Float16* Vh  = (_Float16*)(ws);             ws += szX;
    _Float16* Oh  = (_Float16*)(ws);             ws += szX;

    // 1) effective weights (mismatch applied) -> f16
    const int nW = E_DIM * E_DIM;
    weff_kernel<<<(nW + 255) / 256, 256, 0, stream>>>(Wq, dq, WqH, nW);
    weff_kernel<<<(nW + 255) / 256, 256, 0, stream>>>(Wk, dk, WkH, nW);
    weff_kernel<<<(nW + 255) / 256, 256, 0, stream>>>(Wv, dv, WvH, nW);
    weff_kernel<<<(nW + 255) / 256, 256, 0, stream>>>(Wo, dO, WoH, nW);

    // 2) activations -> f16
    const int nX = M_DIM * E_DIM;
    f32_to_f16_kernel<<<(nX + 255) / 256, 256, 0, stream>>>(xq, XqH, nX);
    f32_to_f16_kernel<<<(nX + 255) / 256, 256, 0, stream>>>(xk, XkH, nX);
    f32_to_f16_kernel<<<(nX + 255) / 256, 256, 0, stream>>>(xv, XvH, nX);

    // 3) analog projections (WMMA + TDM staging); Q carries scale D^-1/2 = 0.125
    const dim3 ggrid(M_DIM / 64, E_DIM / 128);
    analog_gemm_kernel<<<ggrid, 256, 0, stream>>>(XqH, WqH, bq, nullptr, Qh, 0.125f, 101u);
    analog_gemm_kernel<<<ggrid, 256, 0, stream>>>(XkH, WkH, bk, nullptr, Kh, 1.0f,   202u);
    analog_gemm_kernel<<<ggrid, 256, 0, stream>>>(XvH, WvH, bv, nullptr, Vh, 1.0f,   303u);

    // 4) attention + head-mean (12 waves/block = 12 heads, deterministic LDS reduce)
    attn_kernel<<<dim3(T_DIM / 16, B_DIM), 384, 0, stream>>>(Qh, Kh, Vh, Oh, attn_mean);

    // 5) analog output projection -> f32 d_out
    analog_gemm_kernel<<<ggrid, 256, 0, stream>>>(Oh, WoH, bo, out, nullptr, 1.0f, 404u);
}
